// CBGNN_my_81484119540343
// MI455X (gfx1250) — matrix operand order repaired and checked by reading, not compiled, not run on hardware
//
#include <hip/hip_runtime.h>
#include <hip/hip_bf16.h>
#include <math.h>

#define N_CYCLES   262144
#define OUT_DIM    256
#define E_CC       4194304
#define M_E2C      4194304
#define LEN_EDGES  1048576
#define NEG_SLOPE  0.2f
#define LN_EPS     1e-5f

typedef __attribute__((ext_vector_type(16))) __bf16 v16bf;
typedef __attribute__((ext_vector_type(8)))  float  v8f;

// key for -infinity under order-preserving float->int mapping
#define KEY_NEGINF ((int)0x807fffff)

__device__ __forceinline__ float leaky(float v) { return v > 0.f ? v : NEG_SLOPE * v; }

// order-preserving float <-> int key (works for +/- values, monotone under int compare)
__device__ __forceinline__ int   fkey(float f)  { int s = __float_as_int(f); return s >= 0 ? s : (s ^ 0x7fffffff); }
__device__ __forceinline__ float fdecode(int k) { int s = (k >= 0) ? k : (k ^ 0x7fffffff); return __int_as_float(s); }

// ---------------------------------------------------------------------------
// K0: pack W1 (256x256 f32, row-major K x N) into bf16 B-fragments.
// Fragment (kt, nt): lane l holds 16 contiguous bf16 = column N = nt*16+(l&15),
// K = kt*32 + (l>=16 ? 8 : 0) + [0..7] and +16+[0..7]  (matches 16-bit B VGPR layout).
// ---------------------------------------------------------------------------
__global__ void k_pack_w1(const float* __restrict__ W1, __bf16* __restrict__ Wp) {
    int tid  = blockIdx.x * blockDim.x + threadIdx.x;   // 4096 total
    int lane = tid & 31;
    int nt   = (tid >> 5) & 15;
    int kt   = tid >> 9;
    int n     = nt * 16 + (lane & 15);
    int kbase = kt * 32 + ((lane >> 4) ? 8 : 0);
    __bf16* dst = Wp + (size_t)tid * 16;
#pragma unroll
    for (int i = 0; i < 16; ++i)
        dst[i] = (__bf16)W1[(size_t)(kbase + (i < 8 ? i : i + 8)) * OUT_DIM + n];
}

// ---------------------------------------------------------------------------
// K1: fused  out1 = ((LN(leaky(x @ W1)) * g1 + b1) @ W2) + b2   per row.
// 64 rows per block, 8 waves: wave w -> row tile (w&3), col half (w>>2).
// ---------------------------------------------------------------------------
__global__ __launch_bounds__(256, 1) void k_mlp1(
    const float* __restrict__ x, const __bf16* __restrict__ Wp,
    const float* __restrict__ g1, const float* __restrict__ b1,
    const float* __restrict__ W2, const float* __restrict__ b2,
    float* __restrict__ out1)
{
    __shared__ float hbuf[64 * 260];          // 66.5 KB, padded pitch

    const int lane    = threadIdx.x & 31;
    const int w       = threadIdx.x >> 5;
    const int rowTile = w & 3;
    const int colHalf = w >> 2;
    const int m       = blockIdx.x * 64 + rowTile * 16 + (lane & 15);
    const int koff    = (lane >> 4) * 8;      // 16-bit A layout: half-wave K offset

    v8f acc[8];
#pragma unroll
    for (int t = 0; t < 8; ++t) acc[t] = (v8f)(0.f);

    const float* xrow = x + (size_t)m * OUT_DIM;

    for (int kt = 0; kt < 8; ++kt) {
        const int k0 = kt * 32;
        // A fragment: lane holds K = k0+koff+[0..7] and k0+16+koff+[0..7]
        float4 c0a = *(const float4*)(xrow + k0 + koff);
        float4 c0b = *(const float4*)(xrow + k0 + koff + 4);
        float4 c1a = *(const float4*)(xrow + k0 + 16 + koff);
        float4 c1b = *(const float4*)(xrow + k0 + 16 + koff + 4);
        if (kt < 7) __builtin_prefetch(xrow + k0 + 32 + koff, 0, 1); // global_prefetch_b8

        float cv[16] = { c0a.x, c0a.y, c0a.z, c0a.w,  c0b.x, c0b.y, c0b.z, c0b.w,
                         c1a.x, c1a.y, c1a.z, c1a.w,  c1b.x, c1b.y, c1b.z, c1b.w };
        v16bf a;
#pragma unroll
        for (int i = 0; i < 16; ++i) a[i] = (__bf16)cv[i];

#pragma unroll
        for (int t = 0; t < 8; ++t) {
            const int nt = colHalf * 8 + t;
            v16bf bfrag = *(const v16bf*)(Wp + ((size_t)(kt * 16 + nt) * 32 + lane) * 16);
            acc[t] = __builtin_amdgcn_wmma_f32_16x16x32_bf16(
                         false, a, false, bfrag, (short)0, acc[t], false, false);
        }
    }

    // leaky + scatter H tile to LDS (C/D layout: vgpr r -> M = r + (lane>=16 ? 8:0))
    const int r0 = rowTile * 16 + ((lane >> 4) ? 8 : 0);
#pragma unroll
    for (int t = 0; t < 8; ++t) {
        const int nbase = colHalf * 128 + t * 16 + (lane & 15);
#pragma unroll
        for (int r = 0; r < 8; ++r)
            hbuf[(r0 + r) * 260 + nbase] = leaky(acc[t][r]);
    }
    __syncthreads();

    // per-row LayerNorm + dot(W2): 4 lanes per row, 64 cols each
    const int row = threadIdx.x >> 2;
    const int q   = threadIdx.x & 3;
    const float* hr = hbuf + row * 260 + q * 64;
    float s = 0.f, ss = 0.f;
#pragma unroll 8
    for (int i = 0; i < 64; ++i) { float v = hr[i]; s += v; ss += v * v; }
    s  += __shfl_xor(s, 1);  s  += __shfl_xor(s, 2);
    ss += __shfl_xor(ss, 1); ss += __shfl_xor(ss, 2);
    const float mu  = s * (1.f / 256.f);
    const float var = ss * (1.f / 256.f) - mu * mu;
    const float rs  = rsqrtf(var + LN_EPS);

    const float* gq = g1 + q * 64;
    const float* bq = b1 + q * 64;
    const float* wq = W2 + q * 64;
    float dot = 0.f;
#pragma unroll 8
    for (int i = 0; i < 64; ++i)
        dot += ((hr[i] - mu) * rs * gq[i] + bq[i]) * wq[i];
    dot += __shfl_xor(dot, 1); dot += __shfl_xor(dot, 2);
    if (q == 0) out1[blockIdx.x * 64 + row] = dot + b2[0];
}

// ---------------------------------------------------------------------------
// K2: init segment buffers
// ---------------------------------------------------------------------------
__global__ void k_init(int* pmaxk, float* denom, float* agg, int* omk) {
    int i = blockIdx.x * blockDim.x + threadIdx.x;
    if (i < N_CYCLES) { pmaxk[i] = KEY_NEGINF; denom[i] = 0.f; agg[i] = 0.f; }
    if (i < LEN_EDGES) omk[i] = KEY_NEGINF;
}

// K3: segment max of leaky(permuteCE) over seg0
__global__ void k_segmax(const int* __restrict__ seg0, const float* __restrict__ pCE,
                         int* __restrict__ pmaxk) {
    int i = blockIdx.x * blockDim.x + threadIdx.x;
    if (i >= E_CC) return;
    atomicMax(&pmaxk[seg0[i]], fkey(leaky(pCE[i])));
}

// K4: denom = segment sum of exp(p - pmax)
__global__ void k_denom(const int* __restrict__ seg0, const float* __restrict__ pCE,
                        const int* __restrict__ pmaxk, float* __restrict__ denom) {
    int i = blockIdx.x * blockDim.x + threadIdx.x;
    if (i >= E_CC) return;
    int s0 = seg0[i];
    float pm = fdecode(pmaxk[s0]);
    if (!isfinite(pm)) pm = 0.f;
    atomicAdd(&denom[s0], __expf(leaky(pCE[i]) - pm));
}

// K5: agg = segment sum of out1[seg1] * attn
__global__ void k_attn(const int* __restrict__ seg0, const int* __restrict__ seg1,
                       const float* __restrict__ pCE, const int* __restrict__ pmaxk,
                       const float* __restrict__ denom, const float* __restrict__ out1,
                       float* __restrict__ agg) {
    int i = blockIdx.x * blockDim.x + threadIdx.x;
    if (i >= E_CC) return;
    int s0 = seg0[i];
    float pm = fdecode(pmaxk[s0]);
    if (!isfinite(pm)) pm = 0.f;
    float e    = __expf(leaky(pCE[i]) - pm);
    float attn = e / (denom[s0] + 1e-16f);
    atomicAdd(&agg[s0], out1[seg1[i]] * attn);
}

// K7: om = segment max of agg[E2C[:,1]] over E2C[:,0]
__global__ void k_om(const int* __restrict__ e2c, const float* __restrict__ agg,
                     int* __restrict__ omk) {
    int i = blockIdx.x * blockDim.x + threadIdx.x;
    if (i >= M_E2C) return;
    int d = e2c[2 * i], s = e2c[2 * i + 1];
    atomicMax(&omk[d], fkey(agg[s]));
}

// ---------------------------------------------------------------------------
// K8: stage-4 MLP, one wave per row (feat = [om, whether_k], K = 2 outer product),
// fused leaky + LN + dot(Wk2); per-block partial sum of squares for L2 norm.
// ---------------------------------------------------------------------------
__global__ __launch_bounds__(256) void k_mlp2(
    const int* __restrict__ omk, const float* __restrict__ wk,
    const float* __restrict__ Wk1, const float* __restrict__ gk,
    const float* __restrict__ bk, const float* __restrict__ Wk2,
    const float* __restrict__ bk2, float* __restrict__ out,
    float* __restrict__ partials)
{
    const int lane = threadIdx.x & 31;
    const int w    = threadIdx.x >> 5;
    const int row  = blockIdx.x * 8 + w;

    float f0 = fdecode(omk[row]);
    if (f0 == -__builtin_inff()) f0 = 0.f;       // isneginf -> 0
    const float f1 = wk[row];

    float h[8];
    float s = 0.f, ss = 0.f;
#pragma unroll
    for (int j = 0; j < 8; ++j) {
        const int c = j * 32 + lane;
        float v = leaky(f0 * Wk1[c] + f1 * Wk1[256 + c]);
        h[j] = v; s += v; ss += v * v;
    }
    for (int off = 16; off; off >>= 1) { s += __shfl_xor(s, off); ss += __shfl_xor(ss, off); }
    const float mu  = s * (1.f / 256.f);
    const float var = ss * (1.f / 256.f) - mu * mu;
    const float rs  = rsqrtf(var + LN_EPS);

    float dot = 0.f;
#pragma unroll
    for (int j = 0; j < 8; ++j) {
        const int c = j * 32 + lane;
        dot += ((h[j] - mu) * rs * gk[c] + bk[c]) * Wk2[c];
    }
    for (int off = 16; off; off >>= 1) dot += __shfl_xor(dot, off);

    __shared__ float pw[8];
    const float o = dot + bk2[0];
    if (lane == 0) { out[row] = o; pw[w] = o * o; }
    __syncthreads();
    if (threadIdx.x == 0) {
        float t = 0.f;
#pragma unroll
        for (int i = 0; i < 8; ++i) t += pw[i];
        partials[blockIdx.x] = t;
    }
}

// K9: single-block reduction of block partials -> normsq
__global__ void k_norm(const float* __restrict__ partials, int n, float* __restrict__ normsq) {
    __shared__ float red[256];
    float s = 0.f;
    for (int i = threadIdx.x; i < n; i += 256) s += partials[i];
    red[threadIdx.x] = s;
    __syncthreads();
    for (int st = 128; st; st >>= 1) {
        if (threadIdx.x < st) red[threadIdx.x] += red[threadIdx.x + st];
        __syncthreads();
    }
    if (threadIdx.x == 0) normsq[0] = red[0];
}

// K10: out = sigmoid(out / max(||out||, 1e-12))
__global__ void k_final(float* __restrict__ out, const float* __restrict__ normsq) {
    int i = blockIdx.x * blockDim.x + threadIdx.x;
    if (i >= LEN_EDGES) return;
    float nrm = fmaxf(sqrtf(normsq[0]), 1e-12f);
    float v = out[i] / nrm;
    out[i] = 1.f / (1.f + __expf(-v));
}

// ---------------------------------------------------------------------------
extern "C" void kernel_launch(void* const* d_in, const int* in_sizes, int n_in,
                              void* d_out, int out_size, void* d_ws, size_t ws_size,
                              hipStream_t stream) {
    const float* x    = (const float*)d_in[0];
    const int*   e2c  = (const int*)d_in[1];
    const int*   seg0 = (const int*)d_in[2];
    const int*   seg1 = seg0 + E_CC;
    const float* pCE  = (const float*)d_in[3];
    const float* wk   = (const float*)d_in[4];
    const float* W1   = (const float*)d_in[6];
    const float* g1   = (const float*)d_in[7];
    const float* b1   = (const float*)d_in[8];
    const float* W2   = (const float*)d_in[9];
    const float* b2   = (const float*)d_in[10];
    const float* Wk1  = (const float*)d_in[11];
    const float* gk   = (const float*)d_in[12];
    const float* bk   = (const float*)d_in[13];
    const float* Wk2  = (const float*)d_in[14];
    const float* bk2  = (const float*)d_in[15];
    float* out = (float*)d_out;

    char* ws = (char*)d_ws;
    size_t off = 0;
    __bf16* Wp    = (__bf16*)(ws + off); off += (size_t)65536 * 2;          // 128 KB
    float*  out1  = (float*)(ws + off);  off += (size_t)N_CYCLES * 4;       // 1 MB
    int*    pmaxk = (int*)(ws + off);    off += (size_t)N_CYCLES * 4;       // 1 MB
    float*  denom = (float*)(ws + off);  off += (size_t)N_CYCLES * 4;       // 1 MB
    float*  agg   = (float*)(ws + off);  off += (size_t)N_CYCLES * 4;       // 1 MB
    int*    omk   = (int*)(ws + off);    off += (size_t)LEN_EDGES * 4;      // 4 MB
    float*  parts = (float*)(ws + off);  off += (size_t)(LEN_EDGES / 8) * 4;// 512 KB
    float*  nsq   = (float*)(ws + off);  off += 256;

    k_pack_w1<<<16, 256, 0, stream>>>(W1, Wp);
    k_mlp1<<<N_CYCLES / 64, 256, 0, stream>>>(x, Wp, g1, b1, W2, b2, out1);
    k_init<<<LEN_EDGES / 256, 256, 0, stream>>>(pmaxk, denom, agg, omk);
    k_segmax<<<E_CC / 256, 256, 0, stream>>>(seg0, pCE, pmaxk);
    k_denom<<<E_CC / 256, 256, 0, stream>>>(seg0, pCE, pmaxk, denom);
    k_attn<<<E_CC / 256, 256, 0, stream>>>(seg0, seg1, pCE, pmaxk, denom, out1, agg);
    k_om<<<M_E2C / 256, 256, 0, stream>>>(e2c, agg, omk);
    k_mlp2<<<LEN_EDGES / 8, 256, 0, stream>>>(omk, wk, Wk1, gk, bk, Wk2, bk2, out, parts);
    k_norm<<<1, 256, 0, stream>>>(parts, LEN_EDGES / 8, nsq);
    k_final<<<LEN_EDGES / 256, 256, 0, stream>>>(out, nsq);
}
